// MultiHeadAttentionIntervention_30502857736489
// MI455X (gfx1250) — matrix operand order repaired and checked by reading, not compiled, not run on hardware
//
#include <hip/hip_runtime.h>

// ---------------------------------------------------------------------------
// MHA forward for MI455X (gfx1250, wave32, WMMA bf16 16x16x32 + TDM)
//   q = relu(x Wq^T + bq); k,v likewise
//   y = softmax(q k^T / sqrt(hd)) v   (per head, flash-style streaming)
//   out = relu(y Wo^T + bo)
// ---------------------------------------------------------------------------

typedef __attribute__((ext_vector_type(16))) __bf16 v16bf;
typedef __attribute__((ext_vector_type(8)))  __bf16 v8bf;
typedef __attribute__((ext_vector_type(8)))  float  v8f;
typedef __attribute__((ext_vector_type(4)))  unsigned int u32x4;
typedef __attribute__((ext_vector_type(8)))  int i32x8;
typedef __attribute__((ext_vector_type(4)))  int i32x4;

#define WMMA_BF16(a, b, c) \
  __builtin_amdgcn_wmma_f32_16x16x32_bf16(false, (a), false, (b), (short)0, (c), false, false)

#if defined(__AMDGCN__) && __has_builtin(__builtin_amdgcn_tensor_load_to_lds) && \
    __has_builtin(__builtin_amdgcn_s_wait_tensorcnt)
#define HAVE_TDM 1
#endif

static __device__ __forceinline__ float bf2f(__bf16 x) {
  unsigned short u = __builtin_bit_cast(unsigned short, x);
  unsigned int v = ((unsigned int)u) << 16;
  return __builtin_bit_cast(float, v);
}
static __device__ __forceinline__ __bf16 f2bf(float f) {
  unsigned int u = __builtin_bit_cast(unsigned int, f);
  u += 0x7FFFu + ((u >> 16) & 1u);           // round-to-nearest-even
  unsigned short h = (unsigned short)(u >> 16);
  return __builtin_bit_cast(__bf16, h);
}
static __device__ __forceinline__ v16bf cat8(v8bf lo, v8bf hi) {
  return __builtin_shufflevector(lo, hi, 0,1,2,3,4,5,6,7,8,9,10,11,12,13,14,15);
}
// reductions across the 16-lane half-wave (rows of a WMMA D tile share a half)
static __device__ __forceinline__ float rowmax16(float v) {
#pragma unroll
  for (int m = 1; m <= 8; m <<= 1) v = fmaxf(v, __shfl_xor(v, m, 32));
  return v;
}
static __device__ __forceinline__ float rowsum16(float v) {
#pragma unroll
  for (int m = 1; m <= 8; m <<= 1) v += __shfl_xor(v, m, 32);
  return v;
}

#define DIMM  2048
#define HDIM  1024
#define S_LEN 4096
#define BATCH 2
#define NHEAD 2

// ---------------------------------------------------------------------------
// Kernel 1: C[M,N] = relu(A[M,K] @ W[N,K]^T + bias[N]), bf16 in/out.
// Block tile 128x128, BK=32, 256 threads = 8 waves (4 x 2 wave grid),
// each wave: 2x4 WMMA 16x16 tiles. Double-buffered LDS, one barrier/iter;
// next tile's global loads are issued before the WMMA burst so their
// latency hides under the matrix ops.
// ---------------------------------------------------------------------------
#define BM 128
#define BN 128
#define BK 32

__global__ __launch_bounds__(256)
void gemm_bias_relu(const __bf16* __restrict__ A, const __bf16* __restrict__ W,
                    const __bf16* __restrict__ bias, __bf16* __restrict__ C,
                    int M, int N, int K) {
  __shared__ __align__(32) __bf16 As[2][BM * BK];   // 2 x 8 KB
  __shared__ __align__(32) __bf16 Ws[2][BN * BK];   // 2 x 8 KB

  const int tid  = threadIdx.x;
  const int wave = tid >> 5, lane = tid & 31;
  const int half = lane >> 4, lr = lane & 15;
  const int bm = blockIdx.y * BM, bn = blockIdx.x * BN;
  const int wm = (wave >> 1) * 32;   // wave row: 4 x 32
  const int wn = (wave & 1) * 64;    // wave col: 2 x 64

  v8f acc[2][4] = {};

  // prologue: stage tile 0
#pragma unroll
  for (int i = 0; i < 2; ++i) {
    int c = tid + i * 256;
    int row = c >> 2, koff = (c & 3) << 3;
    *(v8bf*)&As[0][row * BK + koff] = *(const v8bf*)&A[(size_t)(bm + row) * K + koff];
    *(v8bf*)&Ws[0][row * BK + koff] = *(const v8bf*)&W[(size_t)(bn + row) * K + koff];
  }
  __syncthreads();

  int p = 0;
  for (int k0 = 0; k0 < K; k0 += BK) {
    const bool has_next = (k0 + BK) < K;
    v8bf pa[2], pw[2];
    if (has_next) {                       // issue next tile's loads early
      int kn = k0 + BK;
#pragma unroll
      for (int i = 0; i < 2; ++i) {
        int c = tid + i * 256;
        int row = c >> 2, koff = (c & 3) << 3;
        pa[i] = *(const v8bf*)&A[(size_t)(bm + row) * K + kn + koff];
        pw[i] = *(const v8bf*)&W[(size_t)(bn + row) * K + kn + koff];
      }
    }

    v16bf af[2], bfr[4];
#pragma unroll
    for (int i = 0; i < 2; ++i) {        // A frag: lane=row lr, K=16*half..+15
      const __bf16* pp = &As[p][(wm + i * 16 + lr) * BK];
      af[i] = *(const v16bf*)(pp + 16 * half);
    }
#pragma unroll
    for (int j = 0; j < 4; ++j)          // B frag: lane=col lr, K=16*half..+15
      bfr[j] = *(const v16bf*)&Ws[p][(wn + j * 16 + lr) * BK + 16 * half];

#pragma unroll
    for (int i = 0; i < 2; ++i)
#pragma unroll
      for (int j = 0; j < 4; ++j)
        acc[i][j] = WMMA_BF16(af[i], bfr[j], acc[i][j]);

    if (has_next) {                       // park next tile in the other buffer
#pragma unroll
      for (int i = 0; i < 2; ++i) {
        int c = tid + i * 256;
        int row = c >> 2, koff = (c & 3) << 3;
        *(v8bf*)&As[p ^ 1][row * BK + koff] = pa[i];
        *(v8bf*)&Ws[p ^ 1][row * BK + koff] = pw[i];
      }
    }
    __syncthreads();                      // single barrier per iteration
    p ^= 1;
  }

  // epilogue: bias + relu; D layout: elem e -> row e+8*half, col lr
#pragma unroll
  for (int i = 0; i < 2; ++i) {
    int mrow = bm + wm + i * 16 + 8 * half;
#pragma unroll
    for (int j = 0; j < 4; ++j) {
      int col = bn + wn + j * 16 + lr;
      float bv = bf2f(bias[col]);
#pragma unroll
      for (int e = 0; e < 8; ++e) {
        float v = acc[i][j][e] + bv;
        v = v > 0.0f ? v : 0.0f;
        C[(size_t)(mrow + e) * N + col] = f2bf(v);
      }
    }
  }
}

// ---------------------------------------------------------------------------
// Kernel 2: flash attention, one (b,h), 128 queries, 128-wide hd slice per
// block. 8 waves; each wave owns 16 query rows (Q frags resident in regs) and
// a 16x128 f32 output accumulator. K streamed in 32-key chunks; the 32x1024B
// K tile (4KB row stride) is fetched by the Tensor Data Mover (D# descriptor,
// wave 0 issues, s_wait_tensorcnt + barrier before use).
// ---------------------------------------------------------------------------
#define KC  32    // keys per chunk
#define DSL 128   // output-dim slice per block

__global__ __launch_bounds__(256)
void flash_attn(const __bf16* __restrict__ Q, const __bf16* __restrict__ Kg,
                const __bf16* __restrict__ V, __bf16* __restrict__ Y) {
  __shared__ __align__(32) __bf16 Ks[KC * 512];       // 32 KB (half of hd)
  __shared__ __align__(32) __bf16 Vt[DSL * KC];       // 8 KB, transposed [col][key]
  __shared__ __align__(32) __bf16 Ps[8][16 * KC];     // 8 KB, per-wave P tile

  const int tid  = threadIdx.x;
  const int wave = tid >> 5, lane = tid & 31;
  const int half = lane >> 4, lr = lane & 15;
  const int bh = blockIdx.z;                 // 0..3
  const int b = bh >> 1, h = bh & 1;
  const int dsl = blockIdx.y * DSL;          // hd slice base
  const int q0 = blockIdx.x * 128;

  const __bf16* Qb = Q  + (size_t)b * S_LEN * DIMM + (size_t)h * HDIM;
  const __bf16* Kb = Kg + (size_t)b * S_LEN * DIMM + (size_t)h * HDIM;
  const __bf16* Vb = V  + (size_t)b * S_LEN * DIMM + (size_t)h * HDIM + dsl;
  __bf16*       Yb = Y  + (size_t)b * S_LEN * DIMM + (size_t)h * HDIM + dsl;

  // preload this wave's 16 query rows as 32 A-fragments (hd = 32 x 32)
  v16bf qf[32];
  {
    const __bf16* qp = Qb + (size_t)(q0 + wave * 16 + lr) * DIMM;
#pragma unroll
    for (int kf = 0; kf < 32; ++kf)
      qf[kf] = *(const v16bf*)(qp + kf * 32 + 16 * half);
  }

  v8f oacc[8] = {};
  float m_run[8], l_run[8];
#pragma unroll
  for (int e = 0; e < 8; ++e) { m_run[e] = -1e30f; l_run[e] = 0.0f; }

  const float scale = 0.03125f;   // 1/sqrt(1024)

  for (int kc = 0; kc < S_LEN; kc += KC) {
    v8f sacc[2] = {};
#pragma unroll
    for (int hh = 0; hh < 2; ++hh) {
#ifdef HAVE_TDM
      // ---- TDM: DMA 32 rows x 1024B (row stride 4096B) into Ks ----
      if (wave == 0) {
        unsigned ldsa = (unsigned)(size_t)(&Ks[0]);
        unsigned long long ga =
            (unsigned long long)(size_t)(Kb + (size_t)kc * DIMM + hh * 512);
        // D# group0: count=1 | lds_addr | global_addr[56:0] | type=2
        u32x4 g0 = { 1u, ldsa, (unsigned)ga,
                     (unsigned)((ga >> 32) & 0x1FFFFFFull) | 0x80000000u };
        // D# group1: data_size=8B units; tensor_dim0/1 = 1<<20 (no OOB clip);
        // tile 128x32 units; tensor_dim0_stride = 512 units (4096B)
        i32x8 g1 = { (3 << 16),              // wg_mask=0, data_size=3 (8B)
                     0,                       // barrier_addr=0 | td0[15:0]<<16
                     16,                      // td0[31:16]=16 | td1[15:0]<<16
                     (128 << 16) | 16,        // td1[31:16]=16 | tile_dim0=128
                     32,                      // tile_dim1=32, tile_dim2=0
                     512, 0, 0 };             // dim0_stride=512, dim1_stride=0
        i32x4 gz = { 0, 0, 0, 0 };
#if defined(__clang_major__) && __clang_major__ >= 23
        i32x8 gz8 = { 0, 0, 0, 0, 0, 0, 0, 0 };
        __builtin_amdgcn_tensor_load_to_lds(g0, g1, gz, gz, gz8, 0);
#else
        __builtin_amdgcn_tensor_load_to_lds(g0, g1, gz, gz, 0);
#endif
        __builtin_amdgcn_s_wait_tensorcnt(0);
      }
#else
      // fallback: manual staging, 2048 8-elem chunks, 8 per thread
#pragma unroll
      for (int i = 0; i < 8; ++i) {
        int c = tid + i * 256;
        int krow = c >> 6, koff = (c & 63) << 3;
        *(v8bf*)&Ks[krow * 512 + koff] =
            *(const v8bf*)&Kb[(size_t)(kc + krow) * DIMM + hh * 512 + koff];
      }
#endif
      if (hh == 0) {
        // stage V chunk transposed: Vt[col][key]; 512 8-col chunks, 2 per thread
#pragma unroll
        for (int i = 0; i < 2; ++i) {
          int c = tid + i * 256;
          int key = c >> 4, colo = (c & 15) << 3;
          v8bf vv = *(const v8bf*)&Vb[(size_t)(kc + key) * DIMM + colo];
#pragma unroll
          for (int e = 0; e < 8; ++e) Vt[(colo + e) * KC + key] = vv[e];
        }
      }
      __syncthreads();
      // scores: two 16-key N tiles, 16 WMMA k-steps over this hd half
#pragma unroll
      for (int nt = 0; nt < 2; ++nt)
#pragma unroll
        for (int kf = 0; kf < 16; ++kf) {
          v16bf kfrag = *(const v16bf*)&Ks[(nt * 16 + lr) * 512 + kf * 32 + 16 * half];
          sacc[nt] = WMMA_BF16(qf[hh * 16 + kf], kfrag, sacc[nt]);
        }
      __syncthreads();
    }

    // online softmax over this 32-key chunk
#pragma unroll
    for (int e = 0; e < 8; ++e) {
      float s0 = sacc[0][e] * scale;
      float s1 = sacc[1][e] * scale;
      float mx = rowmax16(fmaxf(s0, s1));
      float mnew = fmaxf(m_run[e], mx);
      float r  = __expf(m_run[e] - mnew);
      float p0 = __expf(s0 - mnew);
      float p1 = __expf(s1 - mnew);
      l_run[e] = l_run[e] * r + rowsum16(p0 + p1);
      m_run[e] = mnew;
#pragma unroll
      for (int j = 0; j < 8; ++j) oacc[j][e] *= r;
      // park P in LDS to re-fragment as WMMA A operand (row e+8*half, key nt*16+lr)
      Ps[wave][(e + 8 * half) * KC + lr]      = f2bf(p0);
      Ps[wave][(e + 8 * half) * KC + 16 + lr] = f2bf(p1);
    }

    // O += P(16x32) x V(32x128)
    v16bf pf = *(const v16bf*)&Ps[wave][lr * KC + 16 * half];
#pragma unroll
    for (int j = 0; j < 8; ++j) {
      v16bf vf = *(const v16bf*)&Vt[(j * 16 + lr) * KC + 16 * half];
      oacc[j] = WMMA_BF16(pf, vf, oacc[j]);
    }
    __syncthreads();   // protect Ks/Vt before next chunk's staging
  }

  // normalize and store this wave's 16 rows x 128 cols
  const int orow = q0 + wave * 16 + 8 * half;
#pragma unroll
  for (int e = 0; e < 8; ++e) {
    float inv = 1.0f / l_run[e];
#pragma unroll
    for (int j = 0; j < 8; ++j)
      Yb[(size_t)(orow + e) * DIMM + j * 16 + lr] = f2bf(oacc[j][e] * inv);
  }
}

// ---------------------------------------------------------------------------
// Launch: 3 projection GEMMs -> flash attention -> output GEMM
// ---------------------------------------------------------------------------
extern "C" void kernel_launch(void* const* d_in, const int* in_sizes, int n_in,
                              void* d_out, int out_size, void* d_ws, size_t ws_size,
                              hipStream_t stream) {
  (void)in_sizes; (void)n_in; (void)out_size; (void)ws_size;

  const __bf16* base = (const __bf16*)d_in[0];
  const __bf16* Wq   = (const __bf16*)d_in[1];
  const __bf16* bq   = (const __bf16*)d_in[2];
  const __bf16* Wk   = (const __bf16*)d_in[3];
  const __bf16* bk   = (const __bf16*)d_in[4];
  const __bf16* Wv   = (const __bf16*)d_in[5];
  const __bf16* bv   = (const __bf16*)d_in[6];
  const __bf16* Wo   = (const __bf16*)d_in[7];
  const __bf16* bo   = (const __bf16*)d_in[8];

  const size_t NE = (size_t)BATCH * S_LEN * DIMM;   // 16,777,216 elems (33.5 MB bf16)
  __bf16* qws = (__bf16*)d_ws;
  __bf16* kws = qws + NE;
  __bf16* vws = kws + NE;
  __bf16* yws = vws + NE;

  const int M = BATCH * S_LEN;     // 8192
  dim3 blk(256);
  dim3 ggemm(DIMM / BN, M / BM);   // (16, 64)

  gemm_bias_relu<<<ggemm, blk, 0, stream>>>(base, Wq, bq, qws, M, DIMM, DIMM);
  gemm_bias_relu<<<ggemm, blk, 0, stream>>>(base, Wk, bk, kws, M, DIMM, DIMM);
  gemm_bias_relu<<<ggemm, blk, 0, stream>>>(base, Wv, bv, vws, M, DIMM, DIMM);

  dim3 gattn(S_LEN / 128, HDIM / DSL, BATCH * NHEAD);   // (32, 8, 4)
  flash_attn<<<gattn, blk, 0, stream>>>(qws, kws, vws, yws);

  gemm_bias_relu<<<ggemm, blk, 0, stream>>>(yws, Wo, bo, (__bf16*)d_out, M, DIMM, DIMM);
}
